// HyperConv_50096498541045
// MI455X (gfx1250) — matrix operand order repaired and checked by reading, not compiled, not run on hardware
//
#include <hip/hip_runtime.h>
#include <hip/hip_bf16.h>

typedef __attribute__((ext_vector_type(16))) _Float16 v16h;
typedef __attribute__((ext_vector_type(8)))  float    v8f;

#define EMB 128
#define NNODE 50000
#define NPRICE 100
#define NCB 50
#define NCM 500
#define NNZ_E 1000000
#define KY_SPLIT 32

// ---------------------------------------------------------------------------
// Prep: s[j] = rowsum(emb[j]), f16 transpose embT[e][j]; pads zero-filled.
// ---------------------------------------------------------------------------
__global__ void rowsum_transpose_kernel(const float* __restrict__ emb,
                                        _Float16* __restrict__ embT,
                                        float* __restrict__ s,
                                        int m, int mp) {
    int j = blockIdx.x * blockDim.x + threadIdx.x;
    if (j >= mp) return;
    if (j < m) {
        const float* row = emb + (size_t)j * EMB;
        float sum = 0.f;
        #pragma unroll 4
        for (int e = 0; e < EMB; ++e) {
            float v = row[e];
            sum += v;
            embT[(size_t)e * mp + j] = (_Float16)v;
        }
        s[j] = sum;
    } else {
        s[j] = 0.f;
        #pragma unroll 4
        for (int e = 0; e < EMB; ++e) embT[(size_t)e * mp + j] = (_Float16)0.f;
    }
}

// ---------------------------------------------------------------------------
// Core: one wave accumulates a 16x128 tile over K-chunks [j0start, M) step
// STRIDE.  All shapes compile-time.  Per chunk: issue all 8 B-tile loads
// first (v16h = 2x b128 each), then build A (exp math hides load latency),
// then 8 WMMAs.  A (16x32 f16) per lane:
//   lanes 0-15 : row = lane,    K = {0..15}
//   lanes 16-31: row = lane-16, K = {16..31}
// B (32x16 f16) per lane: col = lane&15, K = 16*(lane>>4)+[0..15] ->
// contiguous v16h from transposed table (pad region zero-filled).
// ---------------------------------------------------------------------------
template <int M, int MP, int STRIDE>
__device__ __forceinline__
void intra_core(const float* __restrict__ arow, const float* __restrict__ s,
                const _Float16* __restrict__ bbase, float mv, int j0start,
                v8f (&acc)[8], float& zsum, float& wsum) {
    const int lane    = threadIdx.x & 31;
    const int halfSel = lane >> 4;
    constexpr bool vec4 = ((M & 3) == 0);

    int j0 = j0start;
    for (; j0 + 32 <= M; j0 += STRIDE) {
        v16h b[8];
        #pragma unroll
        for (int c = 0; c < 8; ++c)
            b[c] = *(const v16h*)(bbase + (size_t)(c * 16) * MP + j0);
        __builtin_prefetch(arow + j0 + STRIDE, 0, 3);

        const int kb = j0 + (halfSel << 4);
        float av[16], sv[16];
        if constexpr (vec4) {
            const float4 s0 = *(const float4*)(s + kb);
            const float4 s1 = *(const float4*)(s + kb + 4);
            const float4 s2 = *(const float4*)(s + kb + 8);
            const float4 s3 = *(const float4*)(s + kb + 12);
            const float4 a0 = *(const float4*)(arow + kb);
            const float4 a1 = *(const float4*)(arow + kb + 4);
            const float4 a2 = *(const float4*)(arow + kb + 8);
            const float4 a3 = *(const float4*)(arow + kb + 12);
            sv[0]=s0.x; sv[1]=s0.y; sv[2]=s0.z; sv[3]=s0.w;
            sv[4]=s1.x; sv[5]=s1.y; sv[6]=s1.z; sv[7]=s1.w;
            sv[8]=s2.x; sv[9]=s2.y; sv[10]=s2.z; sv[11]=s2.w;
            sv[12]=s3.x; sv[13]=s3.y; sv[14]=s3.z; sv[15]=s3.w;
            av[0]=a0.x; av[1]=a0.y; av[2]=a0.z; av[3]=a0.w;
            av[4]=a1.x; av[5]=a1.y; av[6]=a1.z; av[7]=a1.w;
            av[8]=a2.x; av[9]=a2.y; av[10]=a2.z; av[11]=a2.w;
            av[12]=a3.x; av[13]=a3.y; av[14]=a3.z; av[15]=a3.w;
        } else {
            #pragma unroll
            for (int t = 0; t < 16; ++t) {
                const int k = kb + t;
                sv[t] = s[k];
                av[t] = arow[k];
            }
        }
        v16h a;
        #pragma unroll
        for (int t = 0; t < 16; ++t) {
            const float e = __expf(mv * sv[t]);
            zsum += e;
            const float w = av[t] * e;
            wsum += w;
            a[t] = (_Float16)w;
        }
        #pragma unroll
        for (int c = 0; c < 8; ++c)
            acc[c] = __builtin_amdgcn_wmma_f32_16x16x32_f16(false, a, false, b[c],
                                                            (short)0, acc[c],
                                                            false, false);
    }

    if constexpr ((M & 31) != 0) {
        if (j0 < M) {  // unique tail chunk: clamped loads + multiplicative mask
            v16h b[8];
            #pragma unroll
            for (int c = 0; c < 8; ++c)
                b[c] = *(const v16h*)(bbase + (size_t)(c * 16) * MP + j0);

            const int kb = j0 + (halfSel << 4);
            v16h a;
            #pragma unroll
            for (int t = 0; t < 16; ++t) {
                const int k  = kb + t;
                const int kc = (k < M) ? k : (M - 1);
                const float e = __expf(mv * s[kc]) * ((k < M) ? 1.f : 0.f);
                zsum += e;
                const float w = arow[kc] * e;
                wsum += w;
                a[t] = (_Float16)w;
            }
            #pragma unroll
            for (int c = 0; c < 8; ++c)
                acc[c] = __builtin_amdgcn_wmma_f32_16x16x32_f16(false, a, false, b[c],
                                                                (short)0, acc[c],
                                                                false, false);
        }
    }
}

// ---------------------------------------------------------------------------
// Direct intra-gate: block = 4 waves, wave w owns row-tile blockIdx.x*4+w
// (16 rows x 128 cols), full K loop, normalized write-out.
// ---------------------------------------------------------------------------
template <int M, int MP>
__global__ __launch_bounds__(128)
void intra_gate_direct_kernel(const float* __restrict__ adj,
                              const float* __restrict__ mat_v,
                              const float* __restrict__ s,
                              const _Float16* __restrict__ embT,
                              float* __restrict__ out, int n) {
    const int lane    = threadIdx.x & 31;
    const int wave    = threadIdx.x >> 5;
    const int halfSel = lane >> 4;
    const int r       = lane & 15;
    const int rowTile = (blockIdx.x * 4 + wave) << 4;
    if (rowTile >= n) return;                       // wave-uniform exit
    const int row = (rowTile + r < n) ? (rowTile + r) : (n - 1);

    const float mv = mat_v[row];
    const float* arow = adj + (size_t)row * (size_t)M;
    const _Float16* bbase = embT + (size_t)r * MP + (halfSel << 4);

    v8f acc[8];
    #pragma unroll
    for (int c = 0; c < 8; ++c) acc[c] = (v8f){};
    float zsum = 0.f, wsum = 0.f;

    intra_core<M, MP, 32>(arow, s, bbase, mv, 0, acc, zsum, wsum);

    zsum += __shfl_xor(zsum, 16);
    wsum += __shfl_xor(wsum, 16);
    const float denom = wsum + 1e-8f * zsum;        // lane l: row l&15

    #pragma unroll
    for (int v = 0; v < 8; ++v) {
        const int lr = v + (halfSel << 3);          // D row of acc[*][v]
        const float ri = 1.0f / __shfl(denom, lr);
        const int orow = rowTile + lr;
        if (orow < n) {
            float* op = out + (size_t)orow * EMB + r;
            #pragma unroll
            for (int c = 0; c < 8; ++c) op[c << 4] = acc[c][v] * ri;
        }
    }
}

// ---------------------------------------------------------------------------
// Split-K intra-gate for wide gates (M = 50000): grid.y = KY_SPLIT K-stripe
// groups, partials accumulated with float atomics into numer/zs/ws.
// ---------------------------------------------------------------------------
template <int M, int MP>
__global__ __launch_bounds__(128)
void intra_gate_splitk_kernel(const float* __restrict__ adj,
                              const float* __restrict__ mat_v,
                              const float* __restrict__ s,
                              const _Float16* __restrict__ embT,
                              float* __restrict__ numer,
                              float* __restrict__ zs, float* __restrict__ ws,
                              int n) {
    const int lane    = threadIdx.x & 31;
    const int wave    = threadIdx.x >> 5;
    const int halfSel = lane >> 4;
    const int r       = lane & 15;
    const int rowTile = blockIdx.x << 4;
    const int j0start = (blockIdx.y * 4 + wave) << 5;
    const int row = (rowTile + r < n) ? (rowTile + r) : (n - 1);

    const float mv = mat_v[row];
    const float* arow = adj + (size_t)row * (size_t)M;
    const _Float16* bbase = embT + (size_t)r * MP + (halfSel << 4);

    v8f acc[8];
    #pragma unroll
    for (int c = 0; c < 8; ++c) acc[c] = (v8f){};
    float zsum = 0.f, wsum = 0.f;

    intra_core<M, MP, KY_SPLIT * 128>(arow, s, bbase, mv, j0start,
                                      acc, zsum, wsum);

    zsum += __shfl_xor(zsum, 16);
    wsum += __shfl_xor(wsum, 16);
    if (lane < 16 && rowTile + lane < n) {          // one add per row per wave
        atomicAdd(&zs[rowTile + lane], zsum);
        atomicAdd(&ws[rowTile + lane], wsum);
    }
    #pragma unroll
    for (int v = 0; v < 8; ++v) {
        const int orow = rowTile + v + (halfSel << 3);
        if (orow < n) {
            float* np = numer + (size_t)orow * EMB + r;
            #pragma unroll
            for (int c = 0; c < 8; ++c) atomicAdd(&np[c << 4], acc[c][v]);
        }
    }
}

__global__ void zero_kernel(float* __restrict__ p, int count) {
    int i = blockIdx.x * blockDim.x + threadIdx.x;
    if (i < count) p[i] = 0.f;
}

__global__ void finalize_splitk_kernel(const float* __restrict__ numer,
                                       const float* __restrict__ zs,
                                       const float* __restrict__ ws,
                                       float* __restrict__ out, int n) {
    int i = blockIdx.x * blockDim.x + threadIdx.x;
    if (i >= n * EMB) return;
    const int rowi = i >> 7;
    out[i] = numer[i] / (ws[rowi] + 1e-8f * zs[rowi]);
}

// ---------------------------------------------------------------------------
// Inter-gate: scores_k = exp(e_k . W + b); out = sum_k (scores_k/tot) * e_k
// ---------------------------------------------------------------------------
__global__ __launch_bounds__(256)
void inter_gate_kernel(const float* __restrict__ W, const float* __restrict__ b,
                       const float* __restrict__ e0, const float* __restrict__ e1,
                       const float* __restrict__ e2, const float* __restrict__ e3,
                       float* __restrict__ out, int n) {
    const int wid  = (int)((blockIdx.x * blockDim.x + threadIdx.x) >> 5);
    const int lane = threadIdx.x & 31;
    if (wid >= n) return;
    const float* es[4] = { e0 + (size_t)wid * EMB, e1 + (size_t)wid * EMB,
                           e2 + (size_t)wid * EMB, e3 + (size_t)wid * EMB };
    float wv[4];
    #pragma unroll
    for (int t = 0; t < 4; ++t) wv[t] = W[lane + 32 * t];

    float sc[4];
    #pragma unroll
    for (int k = 0; k < 4; ++k) {
        float d = 0.f;
        #pragma unroll
        for (int t = 0; t < 4; ++t) d += es[k][lane + 32 * t] * wv[t];
        #pragma unroll
        for (int off = 16; off > 0; off >>= 1) d += __shfl_xor(d, off);
        sc[k] = __expf(d + b[0]);
    }
    const float rt = 1.0f / (sc[0] + sc[1] + sc[2] + sc[3]);
    #pragma unroll
    for (int t = 0; t < 4; ++t) {
        const int c = lane + 32 * t;
        float o = 0.f;
        #pragma unroll
        for (int k = 0; k < 4; ++k) o += (sc[k] * rt) * es[k][c];
        out[(size_t)wid * EMB + c] = o;
    }
}

// ---------------------------------------------------------------------------
// COO SpMM accumulate: out[row] += val * emb[col].  One wave per edge.
// ---------------------------------------------------------------------------
__global__ __launch_bounds__(256)
void spmm_atomic_kernel(const int* __restrict__ rows, const int* __restrict__ cols,
                        const float* __restrict__ vals, const float* __restrict__ emb,
                        float* __restrict__ out, int nnz) {
    const int e    = (int)((blockIdx.x * blockDim.x + threadIdx.x) >> 5);
    const int lane = threadIdx.x & 31;
    if (e >= nnz) return;
    const int   rIdx = rows[e], cIdx = cols[e];
    const float v    = vals[e];
    const float* src = emb + (size_t)cIdx * EMB;
    float*       dst = out + (size_t)rIdx * EMB;
    #pragma unroll
    for (int t = 0; t < 4; ++t)
        atomicAdd(&dst[lane + 32 * t], v * src[lane + 32 * t]);
}

// ---------------------------------------------------------------------------
struct SplitScratch { float* numer; float* zs; float* ws; };

template <int M, int MP>
static inline void launch_direct(const float* adj, const float* mv, const float* s,
                                 const _Float16* embT, float* out, int n,
                                 hipStream_t st) {
    intra_gate_direct_kernel<M, MP><<<dim3((n + 63) / 64), 128, 0, st>>>(
        adj, mv, s, embT, out, n);
}

template <int M, int MP>
static inline void launch_splitk(const float* adj, const float* mv, const float* s,
                                 const _Float16* embT, float* out, int n,
                                 const SplitScratch& sc, hipStream_t st) {
    const int zcount = n * EMB + 2 * n;
    zero_kernel<<<(zcount + 255) / 256, 256, 0, st>>>(sc.numer, zcount);
    dim3 grid((n + 15) / 16, KY_SPLIT);
    intra_gate_splitk_kernel<M, MP><<<grid, 128, 0, st>>>(
        adj, mv, s, embT, sc.numer, sc.zs, sc.ws, n);
    finalize_splitk_kernel<<<(n * EMB + 255) / 256, 256, 0, st>>>(
        sc.numer, sc.zs, sc.ws, out, n);
}

extern "C" void kernel_launch(void* const* d_in, const int* in_sizes, int n_in,
                              void* d_out, int out_size, void* d_ws, size_t ws_size,
                              hipStream_t stream) {
    (void)in_sizes; (void)n_in; (void)out_size;

    const int* adj_rows = (const int*)d_in[0];
    const int* adj_cols = adj_rows + NNZ_E;
    const float* adj_vals = (const float*)d_in[1];

    constexpr int MP_ITEM = 50016, MP_PRI = 128, MP_CB = 64, MP_CM = 512;

    size_t off = 0;
    auto carve = [&](size_t bytes) {
        void* p = (char*)d_ws + off;
        off += (bytes + 255) & ~(size_t)255;
        return p;
    };
    float* s_item = (float*)carve((size_t)MP_ITEM * 4);
    float* s_pri  = (float*)carve((size_t)MP_PRI  * 4);
    float* s_cb   = (float*)carve((size_t)MP_CB   * 4);
    float* s_cm   = (float*)carve((size_t)MP_CM   * 4);
    _Float16* itemT = (_Float16*)carve((size_t)EMB * MP_ITEM * 2);
    _Float16* priT  = (_Float16*)carve((size_t)EMB * MP_PRI  * 2);
    _Float16* cbT   = (_Float16*)carve((size_t)EMB * MP_CB   * 2);
    _Float16* cmT   = (_Float16*)carve((size_t)EMB * MP_CM   * 2);
    float* buf0 = (float*)carve((size_t)NNODE * EMB * 4);
    float* buf1 = (float*)carve((size_t)NNODE * EMB * 4);
    float* buf2 = (float*)carve((size_t)NNODE * EMB * 4);
    float* itemA = (float*)carve((size_t)NNODE  * EMB * 4);
    float* priA  = (float*)carve((size_t)NPRICE * EMB * 4);
    float* cbA   = (float*)carve((size_t)NCB    * EMB * 4);
    float* cmA   = (float*)carve((size_t)NCM    * EMB * 4);
    float* sk = (float*)carve((size_t)NCM * (EMB + 2) * 4);
    SplitScratch ssc{ sk, sk + (size_t)NCM * EMB, sk + (size_t)NCM * EMB + NCM };
    if (off > ws_size) return;

    for (int L = 0; L < 2; ++L) {
        const float* itemC = L ? itemA : (const float*)d_in[14];
        const float* priC  = L ? priA  : (const float*)d_in[15];
        const float* cbC   = L ? cbA   : (const float*)d_in[16];
        const float* cmC   = L ? cmA   : (const float*)d_in[17];
        float* itemD = L ? (float*)d_out                       : itemA;
        float* priD  = L ? (float*)d_out + (size_t)NNODE * EMB : priA;

        rowsum_transpose_kernel<<<(MP_ITEM + 255) / 256, 256, 0, stream>>>(itemC, itemT, s_item, NNODE,  MP_ITEM);
        rowsum_transpose_kernel<<<(MP_PRI  + 255) / 256, 256, 0, stream>>>(priC,  priT,  s_pri,  NPRICE, MP_PRI);
        rowsum_transpose_kernel<<<(MP_CB   + 255) / 256, 256, 0, stream>>>(cbC,   cbT,   s_cb,   NCB,    MP_CB);
        rowsum_transpose_kernel<<<(MP_CM   + 255) / 256, 256, 0, stream>>>(cmC,   cmT,   s_cm,   NCM,    MP_CM);

        // ---- item target -------------------------------------------------
        launch_direct<NPRICE, MP_PRI>((const float*)d_in[3],  (const float*)d_in[18], s_pri, priT, buf0, NNODE, stream);
        launch_direct<NCB,    MP_CB >((const float*)d_in[7],  (const float*)d_in[19], s_cb,  cbT,  buf1, NNODE, stream);
        launch_direct<NCM,    MP_CM >((const float*)d_in[11], (const float*)d_in[20], s_cm,  cmT,  buf2, NNODE, stream);
        inter_gate_kernel<<<(NNODE + 7) / 8, 256, 0, stream>>>(
            (const float*)d_in[30], (const float*)d_in[31], itemC, buf0, buf1, buf2, itemD, NNODE);
        spmm_atomic_kernel<<<(NNZ_E + 7) / 8, 256, 0, stream>>>(
            adj_rows, adj_cols, adj_vals, itemC, itemD, NNZ_E);

        // ---- price target ------------------------------------------------
        launch_splitk<NNODE, MP_ITEM>((const float*)d_in[2], (const float*)d_in[21], s_item, itemT, buf0, NPRICE, ssc, stream);
        launch_direct<NCB,   MP_CB  >((const float*)d_in[4], (const float*)d_in[22], s_cb,   cbT,   buf1, NPRICE, stream);
        launch_direct<NCM,   MP_CM  >((const float*)d_in[8], (const float*)d_in[23], s_cm,   cmT,   buf2, NPRICE, stream);
        inter_gate_kernel<<<(NPRICE + 7) / 8, 256, 0, stream>>>(
            (const float*)d_in[32], (const float*)d_in[33], priC, buf0, buf1, buf2, priD, NPRICE);

        if (L == 0) {  // cb/cm updates feed only the next layer
            launch_direct<NPRICE, MP_PRI >((const float*)d_in[5],  (const float*)d_in[24], s_pri,  priT,  buf0, NCB, stream);
            launch_splitk<NNODE,  MP_ITEM>((const float*)d_in[6],  (const float*)d_in[25], s_item, itemT, buf1, NCB, ssc, stream);
            launch_direct<NCM,    MP_CM  >((const float*)d_in[12], (const float*)d_in[26], s_cm,   cmT,   buf2, NCB, stream);
            inter_gate_kernel<<<(NCB + 7) / 8, 256, 0, stream>>>(
                (const float*)d_in[34], (const float*)d_in[35], cbC, buf0, buf1, buf2, cbA, NCB);

            launch_direct<NPRICE, MP_PRI >((const float*)d_in[9],  (const float*)d_in[27], s_pri,  priT,  buf0, NCM, stream);
            launch_splitk<NNODE,  MP_ITEM>((const float*)d_in[10], (const float*)d_in[28], s_item, itemT, buf1, NCM, ssc, stream);
            launch_direct<NCB,    MP_CB  >((const float*)d_in[13], (const float*)d_in[29], s_cb,   cbT,   buf2, NCM, stream);
            inter_gate_kernel<<<(NCM + 7) / 8, 256, 0, stream>>>(
                (const float*)d_in[36], (const float*)d_in[37], cmC, buf0, buf1, buf2, cmA, NCM);
        }
    }
}
